// HazeAttn_50440095924315
// MI455X (gfx1250) — matrix operand order repaired and checked by reading, not compiled, not run on hardware
//
#include <hip/hip_runtime.h>

// ---------------------------------------------------------------------------
// HazeAttn for MI455X (gfx1250, wave32, WMMA + Tensor Data Mover).
// Pipeline: f32->f16 convert | QKV GEMMs (WMMA, TDM-staged A) | rel-score GEMV |
//           fused 2-stream flash attention (WMMA, TDM-staged Q/K) |
//           output GEMM (WMMA, f32 out)
// ---------------------------------------------------------------------------

typedef __attribute__((ext_vector_type(16))) _Float16 v16h;
typedef __attribute__((ext_vector_type(8)))  _Float16 v8h;
typedef __attribute__((ext_vector_type(8)))  float    v8f;
typedef __attribute__((ext_vector_type(4)))  unsigned int u32x4;
typedef __attribute__((ext_vector_type(8)))  int      i32x8;
typedef __attribute__((ext_vector_type(4)))  int      i32x4;

#define WMMA_F16(a, b, c) \
  __builtin_amdgcn_wmma_f32_16x16x32_f16(false, (a), false, (b), (short)0, (c), false, false)

constexpr int Bb = 8, Tt = 1024, Ee = 1024, Hh = 16, Dd = 64, HD = 1024;

// ---------------- Tensor Data Mover (guarded, with manual fallback) --------
#if defined(__gfx1250__) && __has_builtin(__builtin_amdgcn_tensor_load_to_lds)
#define HAVE_TDM 1
#if __has_include(<hip/amd_detail/amd_gfx1250_TDM.h>)
// amdgpu-toolchain (clang-23 / therock headers): 6-arg builtin
#define TDM_CALL(g0, g1, g2, g3) \
  __builtin_amdgcn_tensor_load_to_lds((g0), (g1), (g2), (g3), (i32x8){0,0,0,0,0,0,0,0}, 0)
#else
// ROCm 7.2 (clang-22): 5-arg builtin
#define TDM_CALL(g0, g1, g2, g3) \
  __builtin_amdgcn_tensor_load_to_lds((g0), (g1), (g2), (g3), 0)
#endif
#else
#define HAVE_TDM 0
#endif

__device__ __forceinline__ void wait_tensorcnt0() {
#if __has_builtin(__builtin_amdgcn_s_wait_tensorcnt)
  __builtin_amdgcn_s_wait_tensorcnt(0);
#else
  asm volatile("s_wait_tensorcnt 0x0" ::: "memory");
#endif
}

#if HAVE_TDM
// 2D f16 tile load, global(row stride `stride` elems) -> LDS with row padding.
// pad codes: interval 0=2DW,1=4,...  amount 0=1DW,1=2,...
__device__ __forceinline__ void tdm_load_2d_f16(uint32_t lds_off, const _Float16* g,
                                                uint32_t td0, uint32_t td1,
                                                uint32_t stride, uint32_t padi,
                                                uint32_t pada) {
  uint64_t ga = (uint64_t)(uintptr_t)g;
  u32x4 g0;
  g0[0] = 1u;                                   // count=1 valid descriptor
  g0[1] = lds_off;                              // LDS byte address
  g0[2] = (uint32_t)ga;                         // global addr [31:0]
  g0[3] = (uint32_t)(ga >> 32) | (2u << 30);    // global addr [56:32] | type=2
  i32x8 g1;
  g1[0] = (int)((1u << 16) | (1u << 20) | (padi << 22) | (pada << 25));
  g1[1] = 0;                                    // atomic barrier addr=0, dim0 lo16=0
  g1[2] = (int)0x4000u;                         // tensor_dim0 = 2^30 (no OOB)
  g1[3] = (int)(0x4000u | (td0 << 16));         // tensor_dim1 = 2^30, tile_dim0
  g1[4] = (int)td1;                             // tile_dim1, tile_dim2=0
  g1[5] = (int)stride;                          // tensor_dim0_stride lo32
  g1[6] = 0;
  g1[7] = 0;
  i32x4 z4 = {0, 0, 0, 0};
  TDM_CALL(g0, g1, z4, z4);
}
#endif

// Low 32 bits of a generic LDS pointer == LDS byte offset (ISA §10.2 aperture map).
__device__ __forceinline__ uint32_t lds_off32(const void* p) {
  return (uint32_t)(uintptr_t)p;
}

__device__ __forceinline__ v8f zero8() {
  v8f z = {0.f, 0.f, 0.f, 0.f, 0.f, 0.f, 0.f, 0.f};
  return z;
}

// Build a 16-half A/B fragment from two 16-byte LDS chunks.
__device__ __forceinline__ v16h ldpair(const _Float16* p0, const _Float16* p1) {
  v8h a = *(const v8h*)p0;
  v8h b = *(const v8h*)p1;
  return __builtin_shufflevector(a, b, 0, 1, 2, 3, 4, 5, 6, 7,
                                 8, 9, 10, 11, 12, 13, 14, 15);
}

// ---------------------------------------------------------------------------
// f32 -> f16 conversion
// ---------------------------------------------------------------------------
__global__ void f32_to_f16_kernel(const float* __restrict__ in,
                                  _Float16* __restrict__ out, int n) {
  int i = blockIdx.x * 256 + threadIdx.x;
  if (i < n) out[i] = (_Float16)in[i];
}

// ---------------------------------------------------------------------------
// r[b,h,t] = (1/sqrt(D)) * sum_e x[b,t,e] * wr[h,e,t]   (one wave per (b,h,t))
// ---------------------------------------------------------------------------
__global__ void rel_score_kernel(const float* __restrict__ x,
                                 const float* __restrict__ wr,
                                 float* __restrict__ r) {
  int wid  = (blockIdx.x * blockDim.x + threadIdx.x) >> 5;
  int lane = threadIdx.x & 31;
  int t = wid & (Tt - 1);
  int h = (wid >> 10) & (Hh - 1);
  int b = wid >> 14;
  if (b >= Bb) return;
  const float* xp = x + ((size_t)(b * Tt + t)) * Ee;
  const float* wp = wr + ((size_t)h * Ee) * Tt + t;
  float s = 0.f;
  for (int e = lane; e < Ee; e += 32) s += xp[e] * wp[(size_t)e * Tt];
#pragma unroll
  for (int off = 16; off; off >>= 1) s += __shfl_xor(s, off, 32);
  if (lane == 0) r[(b * Hh + h) * Tt + t] = s * 0.125f;
}

// ---------------------------------------------------------------------------
// WMMA GEMM: C[M,N] = A[M,K] * B[K,N], A/B f16 row-major, f32 accum.
// Block 256 thr (8 waves), tile 128x128, BK=32; wave tile 64x32 (4x2 frags).
// A tile staged by the TDM (row pad 16B -> LDA=40), B staged+transposed by VALU.
// ---------------------------------------------------------------------------
template <bool OUT_F16>
__global__ void __launch_bounds__(256)
gemm_wmma_kernel(const _Float16* __restrict__ A, const _Float16* __restrict__ Bm,
                 void* __restrict__ Cv, int M, int N, int K) {
  constexpr int LDA = 40, LDB = 40;  // +8 halfs pad, keeps 16B alignment
  __shared__ __align__(16) _Float16 As[128 * LDA];
  __shared__ __align__(16) _Float16 Bs[128 * LDB];  // stored [n][k]

  const int tid = threadIdx.x;
  const int lane = tid & 31;
  const int wid = tid >> 5;
  const int half_lane = lane & 15;
  const int hi = lane >> 4;
  const int hi8 = hi * 8, hi16 = hi * 16;
  const int m0 = blockIdx.y * 128, n0 = blockIdx.x * 128;
  const int wm = (wid & 1) * 64;   // wave row base in tile
  const int wn = (wid >> 1) * 32;  // wave col base in tile

  v8f acc[4][2];
#pragma unroll
  for (int i = 0; i < 4; ++i)
#pragma unroll
    for (int j = 0; j < 2; ++j) acc[i][j] = zero8();

  const uint32_t ldsAs = lds_off32(As);

  for (int k0 = 0; k0 < K; k0 += 32) {
    // ---- stage A tile 128x32 ----
#if HAVE_TDM
    if (wid == 0) {
      // 128 rows x 32 halfs(64B), +16B pad/row: interval=16DW(code 3), amount=4DW(code 3)
      tdm_load_2d_f16(ldsAs, A + (size_t)m0 * K + k0, 32u, 128u, (uint32_t)K, 3u, 3u);
    }
#else
    {
      int row = tid >> 1, kofs = (tid & 1) * 16;
      const v8h* ag = (const v8h*)(A + (size_t)(m0 + row) * K + k0 + kofs);
      v8h a0 = ag[0], a1 = ag[1];
      v8h* ad = (v8h*)&As[row * LDA + kofs];
      ad[0] = a0;
      ad[1] = a1;
    }
#endif
    // ---- stage B tile 32x128, transposed to Bs[n][k] ----
    {
      int kk = tid >> 3, nofs = (tid & 7) * 16;
      const v8h* bg = (const v8h*)(Bm + (size_t)(k0 + kk) * N + n0 + nofs);
      v8h b0 = bg[0], b1 = bg[1];
      if (k0 + 32 < K)  // global_prefetch_b8 for next K block of B
        __builtin_prefetch(Bm + (size_t)(k0 + 32 + kk) * N + n0 + nofs, 0, 1);
#pragma unroll
      for (int i = 0; i < 8; ++i) {
        Bs[(nofs + i) * LDB + kk]     = b0[i];
        Bs[(nofs + 8 + i) * LDB + kk] = b1[i];
      }
    }
#if HAVE_TDM
    if (wid == 0) wait_tensorcnt0();
#endif
    __syncthreads();

    // ---- fragments + 8 WMMAs ----
    v16h af[4], bf[2];
#pragma unroll
    for (int mf = 0; mf < 4; ++mf) {
      const _Float16* ap = &As[(wm + mf * 16 + half_lane) * LDA];
      af[mf] = ldpair(ap + hi8, ap + 16 + hi8);
    }
#pragma unroll
    for (int nf = 0; nf < 2; ++nf) {
      const _Float16* bp = &Bs[(wn + nf * 16 + half_lane) * LDB];
      bf[nf] = ldpair(bp + hi8, bp + 16 + hi8);
    }
#pragma unroll
    for (int mf = 0; mf < 4; ++mf)
#pragma unroll
      for (int nf = 0; nf < 2; ++nf) acc[mf][nf] = WMMA_F16(af[mf], bf[nf], acc[mf][nf]);
    __syncthreads();
  }

  // ---- store C (C-layout: VGPR r -> row hi*8+r, lane -> col) ----
#pragma unroll
  for (int mf = 0; mf < 4; ++mf)
#pragma unroll
    for (int nf = 0; nf < 2; ++nf)
#pragma unroll
      for (int rr = 0; rr < 8; ++rr) {
        int row = m0 + wm + mf * 16 + hi8 + rr;
        int col = n0 + wn + nf * 16 + half_lane;
        float v = acc[mf][nf][rr];
        if constexpr (OUT_F16)
          ((_Float16*)Cv)[(size_t)row * N + col] = (_Float16)v;
        else
          ((float*)Cv)[(size_t)row * N + col] = v;
      }
}

// ---------------------------------------------------------------------------
// Online-softmax update for one stream (content or relative).
// ---------------------------------------------------------------------------
__device__ __forceinline__ void stream_update(float lg[4][8], float m[8], float l[8],
                                              v8f acc[4], _Float16* PsW,
                                              const _Float16* Vs, int half_lane, int hi) {
  constexpr int LD = 72;
  const int hi8 = hi * 8, hi16 = hi * 16;
  float scl[8];
#pragma unroll
  for (int rr = 0; rr < 8; ++rr) {
    float mv = -3.0e38f;
#pragma unroll
    for (int f = 0; f < 4; ++f) mv = fmaxf(mv, lg[f][rr]);
#pragma unroll
    for (int off = 1; off < 16; off <<= 1) mv = fmaxf(mv, __shfl_xor(mv, off, 16));
    float mn = fmaxf(m[rr], mv);
    float sc2 = __expf(m[rr] - mn);
    float rs = 0.f;
#pragma unroll
    for (int f = 0; f < 4; ++f) {
      float p = __expf(lg[f][rr] - mn);
      rs += p;
      PsW[(hi8 + rr) * LD + f * 16 + half_lane] = (_Float16)p;
    }
#pragma unroll
    for (int off = 1; off < 16; off <<= 1) rs += __shfl_xor(rs, off, 16);
    l[rr] = l[rr] * sc2 + rs;
    m[rr] = mn;
    scl[rr] = sc2;
  }
#pragma unroll
  for (int df = 0; df < 4; ++df)
#pragma unroll
    for (int rr = 0; rr < 8; ++rr) acc[df][rr] *= scl[rr];

  asm volatile("" ::: "memory");  // keep LDS P writes before the frag reloads

  const _Float16* pp = &PsW[half_lane * LD];
  v16h pa0 = ldpair(pp + hi8, pp + 16 + hi8);
  v16h pa1 = ldpair(pp + 32 + hi8, pp + 48 + hi8);
#pragma unroll
  for (int df = 0; df < 4; ++df) {
    const _Float16* vp = &Vs[(df * 16 + half_lane) * LD];
    v16h vb0 = ldpair(vp + hi8, vp + 16 + hi8);
    v16h vb1 = ldpair(vp + 32 + hi8, vp + 48 + hi8);
    acc[df] = WMMA_F16(pa0, vb0, acc[df]);
    acc[df] = WMMA_F16(pa1, vb1, acc[df]);
  }
}

// ---------------------------------------------------------------------------
// Fused causal attention, two-stream (content + relative), flash-style.
// Grid: (T/64, H, B). Block 128 (4 waves); wave w owns q rows [w*16, w*16+16).
// Q/K tiles staged by TDM (pad to LD=72); V transposed manually into [d][s].
// ---------------------------------------------------------------------------
__global__ void __launch_bounds__(128)
attn_kernel(const _Float16* __restrict__ Q, const _Float16* __restrict__ Kx,
            const _Float16* __restrict__ Vx, const float* __restrict__ rbuf,
            const float* __restrict__ alpha, _Float16* __restrict__ Oa) {
  constexpr int LD = 72;
  __shared__ __align__(16) _Float16 Qs[64 * LD];
  __shared__ __align__(16) _Float16 Ks[64 * LD];      // [s][d]
  __shared__ __align__(16) _Float16 Vs[64 * LD];      // transposed [d][s]
  __shared__ __align__(16) _Float16 Ps[4 * 16 * LD];  // per-wave P staging

  const int tid = threadIdx.x;
  const int lane = tid & 31, w = tid >> 5;
  const int half_lane = lane & 15, hi = lane >> 4;
  const int hi8 = hi * 8, hi16 = hi * 16;
  const int qb = blockIdx.x, h = blockIdx.y, b = blockIdx.z;

  const size_t baseK = ((size_t)(b * Tt)) * HD + h * Dd;

  // ---- stage Q tile (64x64): TDM with row pad (128B data + 16B pad) ----
#if HAVE_TDM
  if (w == 0) {
    tdm_load_2d_f16(lds_off32(Qs), Q + baseK + (size_t)(qb * 64) * HD,
                    64u, 64u, (uint32_t)HD, 4u, 3u);
    wait_tensorcnt0();
  }
#else
  {
    int row = tid >> 1, dofs = (tid & 1) * 32;
    const v8h* src = (const v8h*)(Q + baseK + (size_t)(qb * 64 + row) * HD + dofs);
    v8h q0 = src[0], q1 = src[1], q2 = src[2], q3 = src[3];
    v8h* dst = (v8h*)&Qs[row * LD + dofs];
    dst[0] = q0; dst[1] = q1; dst[2] = q2; dst[3] = q3;
  }
#endif
  __syncthreads();

  v16h qa[2];
  {
    const _Float16* qp = &Qs[(w * 16 + half_lane) * LD];
    qa[0] = ldpair(qp + hi8, qp + 16 + hi8);
    qa[1] = ldpair(qp + 32 + hi8, qp + 48 + hi8);
  }

  float m_ca[8], l_ca[8], m_ra[8], l_ra[8];
  v8f acc_ca[4], acc_ra[4];
#pragma unroll
  for (int rr = 0; rr < 8; ++rr) {
    m_ca[rr] = -3.0e38f; l_ca[rr] = 0.f;
    m_ra[rr] = -3.0e38f; l_ra[rr] = 0.f;
  }
#pragma unroll
  for (int df = 0; df < 4; ++df) { acc_ca[df] = zero8(); acc_ra[df] = zero8(); }

  const float* rp = rbuf + (size_t)(b * Hh + h) * Tt;
  _Float16* PsW = &Ps[w * 16 * LD];
  const int tRow0 = qb * 64 + w * 16 + hi8;
  const uint32_t ldsKs = lds_off32(Ks);

  for (int j = 0; j <= qb; ++j) {
    __syncthreads();  // previous-iteration readers done with Ks/Vs
    // ---- K tile via TDM; V tile manual transpose into Vs[d][s] ----
#if HAVE_TDM
    if (w == 0)
      tdm_load_2d_f16(ldsKs, Kx + baseK + (size_t)(j * 64) * HD, 64u, 64u,
                      (uint32_t)HD, 4u, 3u);
#endif
    {
      int row = tid >> 1, dofs = (tid & 1) * 32;
#if !HAVE_TDM
      const v8h* ksrc = (const v8h*)(Kx + baseK + (size_t)(j * 64 + row) * HD + dofs);
      v8h k0 = ksrc[0], k1 = ksrc[1], k2 = ksrc[2], k3 = ksrc[3];
      v8h* kdst = (v8h*)&Ks[row * LD + dofs];
      kdst[0] = k0; kdst[1] = k1; kdst[2] = k2; kdst[3] = k3;
#endif
      const v8h* vsrc = (const v8h*)(Vx + baseK + (size_t)(j * 64 + row) * HD + dofs);
      v8h v0 = vsrc[0], v1 = vsrc[1], v2 = vsrc[2], v3 = vsrc[3];
#pragma unroll
      for (int i = 0; i < 8; ++i) {  // transpose into Vs[d][s]
        Vs[(dofs + i) * LD + row]      = v0[i];
        Vs[(dofs + 8 + i) * LD + row]  = v1[i];
        Vs[(dofs + 16 + i) * LD + row] = v2[i];
        Vs[(dofs + 24 + i) * LD + row] = v3[i];
      }
    }
#if HAVE_TDM
    if (w == 0) wait_tensorcnt0();
#endif
    __syncthreads();

    // ---- S = Q K^T (16x64 per wave, 8 WMMAs) ----
    v8f sfr[4];
#pragma unroll
    for (int f = 0; f < 4; ++f) {
      const _Float16* kp = &Ks[(f * 16 + half_lane) * LD];
      v16h b0 = ldpair(kp + hi8, kp + 16 + hi8);
      v16h b1 = ldpair(kp + 32 + hi8, kp + 48 + hi8);
      v8f c = zero8();
      c = WMMA_F16(qa[0], b0, c);
      c = WMMA_F16(qa[1], b1, c);
      sfr[f] = c;
    }

    // ---- masked logits for both streams ----
    float lg_ca[4][8], lg_ra[4][8];
#pragma unroll
    for (int f = 0; f < 4; ++f) {
      float rv = rp[j * 64 + f * 16 + half_lane];
      int sCol = j * 64 + f * 16 + half_lane;
#pragma unroll
      for (int rr = 0; rr < 8; ++rr) {
        bool ok = sCol <= (tRow0 + rr);
        float sv = sfr[f][rr] * 0.125f;  // 1/sqrt(D)
        lg_ca[f][rr] = ok ? sv : -3.0e38f;
        lg_ra[f][rr] = ok ? rv : -3.0e38f;
      }
    }

    stream_update(lg_ca, m_ca, l_ca, acc_ca, PsW, Vs, half_lane, hi);
    stream_update(lg_ra, m_ra, l_ra, acc_ra, PsW, Vs, half_lane, hi);
  }

  // ---- blend + write [B,T,H*D] as f16 ----
  float a_sig;
  { float al = alpha[h]; a_sig = 1.f / (1.f + __expf(-al)); }
#pragma unroll
  for (int df = 0; df < 4; ++df)
#pragma unroll
    for (int rr = 0; rr < 8; ++rr) {
      int t = qb * 64 + w * 16 + hi8 + rr;
      int d = df * 16 + half_lane;
      float o = a_sig * (acc_ra[df][rr] / l_ra[rr]) +
                (1.f - a_sig) * (acc_ca[df][rr] / l_ca[rr]);
      Oa[((size_t)(b * Tt) + t) * HD + h * Dd + d] = (_Float16)o;
    }
}

// ---------------------------------------------------------------------------
// launch
// ---------------------------------------------------------------------------
extern "C" void kernel_launch(void* const* d_in, const int* in_sizes, int n_in,
                              void* d_out, int out_size, void* d_ws, size_t ws_size,
                              hipStream_t stream) {
  (void)in_sizes; (void)n_in; (void)out_size; (void)ws_size;
  const float* x     = (const float*)d_in[0];
  const float* wq    = (const float*)d_in[1];
  const float* wk    = (const float*)d_in[2];
  const float* wv    = (const float*)d_in[3];
  const float* wr    = (const float*)d_in[4];
  const float* alpha = (const float*)d_in[5];
  const float* wo    = (const float*)d_in[6];
  float* out = (float*)d_out;

  char* ws = (char*)d_ws;
  size_t off = 0;
  auto alloc = [&](size_t bytes) -> void* {
    off = (off + 255) & ~(size_t)255;
    void* p = ws + off;
    off += bytes;
    return p;
  };

  const size_t nX = (size_t)Bb * Tt * Ee;   // 8388608
  const size_t nW = (size_t)Ee * HD;        // 1048576
  _Float16* xh  = (_Float16*)alloc(nX * 2);
  _Float16* wqh = (_Float16*)alloc(nW * 2);
  _Float16* wkh = (_Float16*)alloc(nW * 2);
  _Float16* wvh = (_Float16*)alloc(nW * 2);
  _Float16* woh = (_Float16*)alloc(nW * 2);
  _Float16* Qh  = (_Float16*)alloc(nX * 2);
  _Float16* Kh  = (_Float16*)alloc(nX * 2);
  _Float16* Vh  = (_Float16*)alloc(nX * 2);
  float*    rb  = (float*)alloc((size_t)Bb * Hh * Tt * 4);
  _Float16* Ah  = (_Float16*)alloc(nX * 2);  // attention output, [B,T,H*D]

  // 1) conversions
  f32_to_f16_kernel<<<(int)(nX / 256), 256, 0, stream>>>(x, xh, (int)nX);
  f32_to_f16_kernel<<<(int)(nW / 256), 256, 0, stream>>>(wq, wqh, (int)nW);
  f32_to_f16_kernel<<<(int)(nW / 256), 256, 0, stream>>>(wk, wkh, (int)nW);
  f32_to_f16_kernel<<<(int)(nW / 256), 256, 0, stream>>>(wv, wvh, (int)nW);
  f32_to_f16_kernel<<<(int)(nW / 256), 256, 0, stream>>>(wo, woh, (int)nW);

  // 2) Q/K/V projections (M=8192, N=1024, K=1024)
  dim3 gg(HD / 128, (Bb * Tt) / 128);
  gemm_wmma_kernel<true><<<gg, 256, 0, stream>>>(xh, wqh, Qh, Bb * Tt, HD, Ee);
  gemm_wmma_kernel<true><<<gg, 256, 0, stream>>>(xh, wkh, Kh, Bb * Tt, HD, Ee);
  gemm_wmma_kernel<true><<<gg, 256, 0, stream>>>(xh, wvh, Vh, Bb * Tt, HD, Ee);

  // 3) relative scores
  rel_score_kernel<<<(Bb * Hh * Tt) / 8, 256, 0, stream>>>(x, wr, rb);

  // 4) fused two-stream causal attention
  dim3 ga(Tt / 64, Hh, Bb);
  attn_kernel<<<ga, 128, 0, stream>>>(Qh, Kh, Vh, rb, alpha, Ah);

  // 5) output projection -> f32 d_out
  gemm_wmma_kernel<false><<<gg, 256, 0, stream>>>(Ah, woh, out, Bb * Tt, HD, Ee);
}
